// GeneratedTreeClassifier_15522011808511
// MI455X (gfx1250) — compile-verified, matrix operand order checked by reading, not run on hardware
//
#include <hip/hip_runtime.h>
#include <hip/hip_bf16.h>

// ---------------------------------------------------------------------------
// Soft decision-tree ensemble, MI455X (gfx1250, wave32, WMMA bf16).
//
// Dominant cost: [65536 x 512] x [512 x Ncols] GEMM. Only internal nodes
// 0..3 of each tree are referenced by the leaf construction (depth-3 quirk
// of the reference), so Ncols = 64 trees * 4 = 256 (not 448).
// Memory-bound: x stream = 134 MB -> ~5.8 us floor @ 23.3 TB/s. Strategy:
// single NT pass over x, packed f32->bf16 convert, LDS-staged A tiles,
// L2-resident prepacked bf16 B (256 KB), v_wmma_f32_16x16x32_bf16 core.
// ---------------------------------------------------------------------------

#define INPUT_DIM  512
#define N_TREES    64
#define N_INTERNAL 7
#define PPT        (N_INTERNAL * (INPUT_DIM + 1) + 16)   // 3607 floats per tree
#define BATCH      65536
#define NCOLS      256                                   // 64 trees * 4 used nodes
#define KSTEPS     (INPUT_DIM / 32)                      // 16
#define XROW       (INPUT_DIM + 8)                       // bf16 row + pad (bank spread)
#define DROW       (NCOLS + 4)                           // f32 decision row + pad

typedef __attribute__((ext_vector_type(16))) __bf16 v16bf;
typedef __attribute__((ext_vector_type(2)))  __bf16 v2bf;
typedef __attribute__((ext_vector_type(2)))  float  v2f;
typedef __attribute__((ext_vector_type(4)))  float  v4f;
typedef __attribute__((ext_vector_type(8)))  float  v8f;
typedef unsigned short u16;

// Packed f32 -> bf16 pair: lowers to hardware packed convert on gfx1250.
static __device__ __forceinline__ unsigned cvt2bf(float a, float b) {
    v2f f = {a, b};
    v2bf r = __builtin_convertvector(f, v2bf);
    return __builtin_bit_cast(unsigned, r);
}

static __device__ __forceinline__ u16 f2bf(float f) {
    unsigned u = __builtin_bit_cast(unsigned, f);
    unsigned r = (u + 0x7FFFu + ((u >> 16) & 1u)) >> 16;   // RNE
    return (u16)r;
}

// ---------------------------------------------------------------------------
// Prep 1: pack split_w (only nodes i<4) into the exact WMMA bf16 B-fragment
// lane layout: Bpack[((ntile*16 + kstep)*32 + lane)*16 + e].
// Lane L holds column n = ntile*16 + (L%16); h = L/16 selects K half-groups:
//   e = 2v+pr -> K = kstep*32 + (v<4 ? 2v : 16+2(v-4)) + 8h + pr
// ---------------------------------------------------------------------------
__global__ void prep_pack_b(const float* __restrict__ tp, u16* __restrict__ bpack) {
    int idx   = blockIdx.x * blockDim.x + threadIdx.x;   // 0 .. 131071
    int e     = idx & 15;
    int lane  = (idx >> 4) & 31;
    int kstep = (idx >> 9) & 15;
    int ntile = idx >> 13;
    int v = e >> 1, pr = e & 1, h = lane >> 4;
    int n = ntile * 16 + (lane & 15);
    int klocal = (v < 4 ? v * 2 : 16 + (v - 4) * 2) + h * 8 + pr;
    int K = kstep * 32 + klocal;
    int t = n >> 2, i = n & 3;                           // tree, internal node (<4)
    bpack[idx] = f2bf(tp[t * PPT + i * INPUT_DIM + K]);
}

// ---------------------------------------------------------------------------
// Prep 2: biases for the 256 used nodes, and Wt[t][l][c] =
// tree_weights[t] * softmax(leaf_logits[t,l,:])[c]  (64*8*2 floats).
// ---------------------------------------------------------------------------
__global__ void prep_misc(const float* __restrict__ tp, const float* __restrict__ tw,
                          float* __restrict__ bias, float* __restrict__ wt) {
    int id = threadIdx.x;                                // 512 threads
    if (id < NCOLS) {
        int t = id >> 2, i = id & 3;
        bias[id] = tp[t * PPT + N_INTERNAL * INPUT_DIM + i];
    }
    int t = id >> 3, l = id & 7;
    const float* ll = tp + t * PPT + N_INTERNAL * (INPUT_DIM + 1) + l * 2;
    float l0 = ll[0], l1 = ll[1];
    float m  = fmaxf(l0, l1);
    float e0 = __expf(l0 - m), e1 = __expf(l1 - m);
    float s  = tw[t] / (e0 + e1);
    wt[t * 16 + l * 2 + 0] = e0 * s;
    wt[t * 16 + l * 2 + 1] = e1 * s;
}

// ---------------------------------------------------------------------------
// Main kernel: 128 threads = 4 waves; each wave owns a 16-row M tile.
// ---------------------------------------------------------------------------
__global__ void __launch_bounds__(128, 1)
forest_kernel(const float* __restrict__ x, const u16* __restrict__ bpack,
              const float* __restrict__ bias, const float* __restrict__ wtg,
              float* __restrict__ out) {
    extern __shared__ char smem[];
    float* wt_lds = (float*)smem;                        // 1024 f32 (4 KB), block-shared
    const int wave = threadIdx.x >> 5;
    const int lane = threadIdx.x & 31;
    char* wbase = smem + 4096 + wave * (XROW * 16 * 2 + DROW * 16 * 4);
    u16*   xt  = (u16*)wbase;                            // 16 x XROW bf16 (A staging)
    float* dec = (float*)(wbase + XROW * 16 * 2);        // 16 x DROW f32 (decisions)

    for (int i = threadIdx.x; i < 1024; i += 128) wt_lds[i] = wtg[i];
    __syncthreads();

    const int mbase = blockIdx.x * 64 + wave * 16;

    // ---- Stage: x tile f32 -> bf16 -> LDS (single NT HBM pass over x) ----
    union Pack { unsigned w[8]; uint4 q[2]; };
    for (int r = 0; r < 16; ++r) {
        const v4f* src = (const v4f*)(x + (size_t)(mbase + r) * INPUT_DIM) + lane * 4;
        Pack p;
#pragma unroll
        for (int q4 = 0; q4 < 4; ++q4) {
            v4f f = __builtin_nontemporal_load(src + q4);      // streaming: keep L2 for B
            p.w[q4 * 2 + 0] = cvt2bf(f.x, f.y);
            p.w[q4 * 2 + 1] = cvt2bf(f.z, f.w);
        }
        uint4* dst = (uint4*)(xt + r * XROW + lane * 16);
        dst[0] = p.q[0];
        dst[1] = p.q[1];
    }
    // Same-wave LDS ordering fence before fragment reads.
    asm volatile("s_wait_dscnt 0" ::: "memory");

    // ---- GEMM: 4 N-blocks x 4 tiles, K = 512 in 16 steps of 32 ----
    const int h  = lane >> 4;
    const int ml = lane & 15;
    union Frag { v16bf bf; uint4 q[2]; };

    for (int nb = 0; nb < 4; ++nb) {
        v8f acc[4] = {v8f{}, v8f{}, v8f{}, v8f{}};
        for (int ks = 0; ks < KSTEPS; ++ks) {
            Frag a;
            const u16* ap = xt + ml * XROW + ks * 32 + h * 8;
            a.q[0] = *(const uint4*)ap;                  // K = ks*32 + 8h + 0..7
            a.q[1] = *(const uint4*)(ap + 16);           // K = ks*32 + 16 + 8h + 0..7
#pragma unroll
            for (int j = 0; j < 4; ++j) {
                Frag b;
                const u16* bp = bpack + (size_t)(((nb * 4 + j) * 16 + ks) * 32 + lane) * 16;
                b.q[0] = *(const uint4*)bp;
                b.q[1] = *(const uint4*)(bp + 8);
                acc[j] = __builtin_amdgcn_wmma_f32_16x16x32_bf16(
                    false, a.bf, false, b.bf, (short)0, acc[j], false, false);
            }
        }
        // ---- Fused bias + sigmoid -> decision LDS ----
#pragma unroll
        for (int j = 0; j < 4; ++j) {
            int n  = (nb * 4 + j) * 16 + ml;
            float bv = bias[n];
#pragma unroll
            for (int r = 0; r < 8; ++r) {
                int m = r + 8 * h;                       // C layout: lanes16-31 -> M+8
                float logit = acc[j][r] + bv;
                dec[m * DROW + n] = 1.0f / (1.0f + __expf(-logit));
            }
        }
    }
    asm volatile("s_wait_dscnt 0" ::: "memory");

    // ---- Tree mixture: lane = (row = lane%16, tree half = lane/16).
    //      Each lane walks 32 trees of one row; single xor-16 reduction. ----
    {
        const int row  = lane & 15;
        const int hsel = lane >> 4;
        const float* drow = dec + row * DROW + hsel * 128;   // 32 trees * 4 decisions
        const float* wrow = wt_lds + hsel * 512;             // 32 trees * 16 weights
        float o0 = 0.f, o1 = 0.f;
#pragma unroll 4
        for (int tt = 0; tt < 32; ++tt) {
            v4f d = *(const v4f*)(drow + tt * 4);
            float inv = 1.0f / (4.0f + d.w + 1e-8f);         // unnormalized leaf-prob sum
            const float* w = wrow + tt * 16;
            float a0 = w[0]  + w[2]  * d.x + w[4]  * (1.f - d.x)
                     + w[6]  * d.y + w[8]  * (1.f - d.y)
                     + w[10] * d.z + w[12] * (1.f - d.z) + w[14] * d.w;
            float a1 = w[1]  + w[3]  * d.x + w[5]  * (1.f - d.x)
                     + w[7]  * d.y + w[9]  * (1.f - d.y)
                     + w[11] * d.z + w[13] * (1.f - d.z) + w[15] * d.w;
            o0 += inv * a0;
            o1 += inv * a1;
        }
        o0 += __shfl_xor(o0, 16, 32);                        // combine tree halves
        o1 += __shfl_xor(o1, 16, 32);
        if (hsel == 0) {
            v2f o = {o0, o1};
            __builtin_nontemporal_store(o, (v2f*)(out + (size_t)(mbase + row) * 2));
        }
    }
}

// ---------------------------------------------------------------------------
extern "C" void kernel_launch(void* const* d_in, const int* in_sizes, int n_in,
                              void* d_out, int out_size, void* d_ws, size_t ws_size,
                              hipStream_t stream) {
    const float* x  = (const float*)d_in[0];
    const float* tp = (const float*)d_in[1];
    const float* tw = (const float*)d_in[2];
    float* out = (float*)d_out;

    u16*   bpack = (u16*)d_ws;                                   // 256 KB
    float* bias  = (float*)((char*)d_ws + 262144);               // 1 KB
    float* wt    = (float*)((char*)d_ws + 262144 + 1024);        // 4 KB

    prep_pack_b<<<512, 256, 0, stream>>>(tp, bpack);
    prep_misc<<<1, 512, 0, stream>>>(tp, tw, bias, wt);

    size_t lds = 4096 + 4 * (size_t)(XROW * 16 * 2 + DROW * 16 * 4); // 137216 B
    static bool attr_set = false;
    if (!attr_set) {
        (void)hipFuncSetAttribute((const void*)forest_kernel,
                                  hipFuncAttributeMaxDynamicSharedMemorySize, (int)lds);
        attr_set = true;
    }
    forest_kernel<<<BATCH / 64, 128, lds, stream>>>(x, bpack, bias, wt, out);
}
